// SCRNALongformer_2164663517722
// MI455X (gfx1250) — matrix operand order repaired and checked by reading, not compiled, not run on hardware
//
#include <hip/hip_runtime.h>
#include <hip/hip_bf16.h>
#include <hip/hip_fp16.h>

// scRNA Longformer forward, MI455X (gfx1250, wave32, WMMA).
// All GEMMs + attention use v_wmma_f32_16x16x32_f16 with f32 accumulation.
// Attention is flash-style fused (BxHxGxG scores would be 512MB >> 192MB L2;
// fused working set ~30MB stays L2 resident). Softmax row-sums are computed
// with a P x ones WMMA; the running max is tile-wide (one 5-step shuffle
// reduction per 32-key chunk). The boolean mask is bit-packed once per launch
// via __ballot. The attention key loop is software-pipelined: next chunk's
// K/V fragments and mask words are fetched under the current chunk's softmax
// VALU work and LDS round trip.

#define DEVINL __device__ __forceinline__

constexpr int B_ = 8, G_ = 2048, D_ = 128, H_ = 4, NCLS_ = 20;
constexpr int DH_ = D_ / H_;   // 32
constexpr int DM_ = 4 * D_;    // 512 MLP hidden
constexpr int BG_ = B_ * G_;   // 16384 rows
constexpr int GW_ = G_ / 32;   // 64 mask words per row

typedef __attribute__((ext_vector_type(16))) _Float16 v16h;
typedef __attribute__((ext_vector_type(8)))  _Float16 v8h;
typedef __attribute__((ext_vector_type(8)))  float    v8f;
typedef __attribute__((ext_vector_type(4)))  float    v4f;

DEVINL v8f wmma_f16(v16h a, v16h b, v8f c) {
  return __builtin_amdgcn_wmma_f32_16x16x32_f16(false, a, false, b, (short)0, c,
                                                false, false);
}

// A-frag: 16x32 f16 from row-major [16 x ld].  ISA layout: lane<16 holds row
// (lane&15), K = {0..7, 16..23}; lane>=16 same row, K = {8..15, 24..31}.
DEVINL v16h load_a(const _Float16* p, int ld) {
  int lane = threadIdx.x & 31;
  const _Float16* row = p + (lane & 15) * ld + (lane >> 4) * 8;
  v8h lo = *(const v8h*)(row);
  v8h hi = *(const v8h*)(row + 16);
  return __builtin_shufflevector(lo, hi, 0,1,2,3,4,5,6,7,8,9,10,11,12,13,14,15);
}

// B-frag: 32x16 f16 from row-major [32 x ld].  Lane holds col (lane&15),
// K = {0..7, 16..23} for lane<16, K = {8..15, 24..31} for lane>=16.
DEVINL v16h load_b(const _Float16* p, int ld) {
  int lane = threadIdx.x & 31;
  const _Float16* col = p + (lane & 15) + (lane >> 4) * 8 * ld;
  v16h r;
#pragma unroll
  for (int s = 0; s < 16; ++s) r[s] = col[((s & 7) + (s >> 3) * 16) * ld];
  return r;
}

// ---------------------------------------------------------------- elementwise
__global__ __launch_bounds__(256) void cast_kernel(const float* __restrict__ in,
                                                   _Float16* __restrict__ out,
                                                   int n) {
  int t = blockIdx.x * 256 + threadIdx.x;
  if (t < n) out[t] = (_Float16)in[t];
}

// Pack bool mask [G,G] (u8) into bit matrix [G, G/32] (u32), bit c = col j0+c.
__global__ __launch_bounds__(256) void packmask_kernel(
    const unsigned char* __restrict__ mask, unsigned int* __restrict__ pm) {
  int t = blockIdx.x * 256 + threadIdx.x;      // < G*G, 32 lanes = 32 cols
  unsigned long long w = __ballot(mask[t] != 0);
  if ((threadIdx.x & 31) == 0) pm[t >> 5] = (unsigned int)w;
}

__global__ __launch_bounds__(256) void embed_kernel(
    const float* __restrict__ values, const float* __restrict__ ge,
    const float* __restrict__ vw, const float* __restrict__ vb,
    float* __restrict__ x) {
  int t = blockIdx.x * 256 + threadIdx.x;   // < BG_*D_
  int d  = t & (D_ - 1);
  int bg = t >> 7;                           // D_=128
  int g  = bg & (G_ - 1);
  x[t] = ge[g * D_ + d] + values[bg] * vw[d] + vb[d];
}

// LayerNorm over D=128, one wave per row, f16 output for WMMA consumption.
__global__ __launch_bounds__(256) void ln_kernel(const float* __restrict__ x,
                                                 const float* __restrict__ gw,
                                                 const float* __restrict__ gb,
                                                 _Float16* __restrict__ out) {
  int row  = blockIdx.x * 8 + (threadIdx.x >> 5);
  int lane = threadIdx.x & 31;
  const float* xr = x + (size_t)row * D_;
  v4f v = *(const v4f*)(xr + lane * 4);
  float s = v[0] + v[1] + v[2] + v[3];
  for (int off = 16; off; off >>= 1) s += __shfl_xor(s, off, 32);
  float mu = s * (1.0f / D_);
  float d0 = v[0] - mu, d1 = v[1] - mu, d2 = v[2] - mu, d3 = v[3] - mu;
  float q = d0 * d0 + d1 * d1 + d2 * d2 + d3 * d3;
  for (int off = 16; off; off >>= 1) q += __shfl_xor(q, off, 32);
  float rs = rsqrtf(q * (1.0f / D_) + 1e-5f);
  int c = lane * 4;
  _Float16* o = out + (size_t)row * D_ + c;
  o[0] = (_Float16)(d0 * rs * gw[c + 0] + gb[c + 0]);
  o[1] = (_Float16)(d1 * rs * gw[c + 1] + gb[c + 1]);
  o[2] = (_Float16)(d2 * rs * gw[c + 2] + gb[c + 2]);
  o[3] = (_Float16)(d3 * rs * gw[c + 3] + gb[c + 3]);
}

// ---------------------------------------------------------------- WMMA GEMM
// C[M x N] = A[M x K](f16) * W[K x N](f16) + bias, fused epilogues.
enum { EPI_ADD_RES = 0, EPI_GELU_F16 = 1, EPI_QV = 2, EPI_KT = 3 };

template <int EPI, int N, int K>
__global__ __launch_bounds__(256) void gemm_kernel(
    const _Float16* __restrict__ A, const _Float16* __restrict__ W,
    const float* __restrict__ bias, float* __restrict__ xres,
    _Float16* __restrict__ outh) {
  constexpr int TN = N / 64;                 // 16x64 tile per wave
  int wave = threadIdx.x >> 5;
  int tile = blockIdx.x * 8 + wave;
  int m0 = (tile / TN) * 16;
  int n0 = (tile % TN) * 64;
  v8f acc[4] = {};
#pragma unroll
  for (int k0 = 0; k0 < K; k0 += 32) {
    v16h a = load_a(A + m0 * K + k0, K);
    acc[0] = wmma_f16(a, load_b(W + k0 * N + n0 +  0, N), acc[0]);
    acc[1] = wmma_f16(a, load_b(W + k0 * N + n0 + 16, N), acc[1]);
    acc[2] = wmma_f16(a, load_b(W + k0 * N + n0 + 32, N), acc[2]);
    acc[3] = wmma_f16(a, load_b(W + k0 * N + n0 + 48, N), acc[3]);
  }
  int lane = threadIdx.x & 31, cl = lane & 15, hi = lane >> 4;
#pragma unroll
  for (int i = 0; i < 4; ++i) {
    int col = n0 + 16 * i + cl;
    float bv = bias[col];
#pragma unroll
    for (int r = 0; r < 8; ++r) {
      int row = m0 + r + hi * 8;             // C-layout: row r / r+8 per half
      float val = acc[i][r] + bv;
      if constexpr (EPI == EPI_ADD_RES) {
        xres[(size_t)row * N + col] += val;
      } else if constexpr (EPI == EPI_GELU_F16) {
        float u = 0.7978845608028654f * (val + 0.044715f * val * val * val);
        outh[(size_t)row * N + col] = (_Float16)(0.5f * val * (1.0f + tanhf(u)));
      } else if constexpr (EPI == EPI_QV) {  // -> [B,H,G,DH]
        int b = row >> 11, g = row & (G_ - 1);
        int h = col >> 5, dh = col & (DH_ - 1);
        outh[(((size_t)b * H_ + h) * G_ + g) * DH_ + dh] = (_Float16)val;
      } else {                               // EPI_KT -> [B,H,DH,G]
        int b = row >> 11, g = row & (G_ - 1);
        int h = col >> 5, dh = col & (DH_ - 1);
        outh[(((size_t)b * H_ + h) * DH_ + dh) * G_ + g] = (_Float16)val;
      }
    }
  }
}

// ------------------------------------------------------- fused attention
// One wave per 16-query tile per (b,h).  q:[B,H,G,DH], kt:[B,H,DH,G],
// v:[B,H,G,DH] f16; pm: bit-packed mask [G, G/32]; o:[B,G,D] f16.
// Software-pipelined: chunk j+1's K/V frags + mask words load under chunk j's
// softmax VALU + LDS round trip.
__global__ __launch_bounds__(256) void attn_kernel(
    const _Float16* __restrict__ q, const _Float16* __restrict__ kt,
    const _Float16* __restrict__ v, const unsigned int* __restrict__ pm,
    _Float16* __restrict__ o) {
  __shared__ _Float16 lp[8][16 * 32];        // per-wave P tile (C->A relayout)
  int wave = threadIdx.x >> 5, lane = threadIdx.x & 31;
  int cl = lane & 15, hi = lane >> 4;
  int t  = blockIdx.x * 8 + wave;            // < B*H*(G/16)
  int qt = t & (G_ / 16 - 1);
  int bh = t >> 7;                           // b*H + h
  int h_ = bh & (H_ - 1), b_ = bh >> 2;
  int q0 = qt * 16;
  const float scale = 0.17677669529663687f;  // DH^-0.5

  v16h qf = load_a(q + ((size_t)bh * G_ + q0) * DH_, DH_);  // full DH=32
  v16h onesb;                                 // B-frag of ones for row-sums
#pragma unroll
  for (int i = 0; i < 16; ++i) onesb[i] = (_Float16)1.0f;

  v8f o0 = {}, o1 = {};
  v8f lrun = {};                              // softmax denominators (C-layout)
  float mrun = -1e30f;                        // tile-wide running max

  const _Float16* ktb = kt + (size_t)bh * DH_ * G_;
  const _Float16* vbp = v + (size_t)bh * G_ * DH_;
  const unsigned int* pmq = pm + (size_t)q0 * GW_;

  // Prologue: fetch chunk 0 fragments and mask words.
  v16h kf0 = load_b(ktb, G_);
  v16h kf1 = load_b(ktb + 16, G_);
  v16h vf0 = load_b(vbp, DH_);
  v16h vf1 = load_b(vbp + 16, DH_);
  unsigned int mw[8];
#pragma unroll
  for (int r = 0; r < 8; ++r) mw[r] = pmq[(r + hi * 8) * GW_];

  for (int j0 = 0; j0 < G_; j0 += 32) {      // 32 keys / iteration
    v8f z = {};
    v8f s0 = wmma_f16(qf, kf0, z);
    v8f s1 = wmma_f16(qf, kf1, z);

    // Issue next chunk's loads now (wraps on last iter; always valid, no
    // branch so EXEC stays all-ones for WMMA).
    int jn = (j0 + 32) & (G_ - 1);
    v16h nkf0 = load_b(ktb + jn, G_);
    v16h nkf1 = load_b(ktb + jn + 16, G_);
    v16h nvf0 = load_b(vbp + (size_t)jn * DH_, DH_);
    v16h nvf1 = load_b(vbp + (size_t)jn * DH_ + 16, DH_);
    unsigned int nmw[8];
#pragma unroll
    for (int r = 0; r < 8; ++r) nmw[r] = pmq[(r + hi * 8) * GW_ + (jn >> 5)];

    float mtile = -1e30f;
#pragma unroll
    for (int r = 0; r < 8; ++r) {            // mask bits + scale + local max
      unsigned int w = mw[r];
      s0[r] = ((w >> cl) & 1u)        ? s0[r] * scale : -1e9f;
      s1[r] = ((w >> (cl + 16)) & 1u) ? s1[r] * scale : -1e9f;
      mtile = fmaxf(mtile, fmaxf(s0[r], s1[r]));
    }
    for (int off = 16; off; off >>= 1)       // 5-op tile-wide max reduction
      mtile = fmaxf(mtile, __shfl_xor(mtile, off, 32));
    float mnew = fmaxf(mrun, mtile);
    float alpha = __expf(mrun - mnew);       // wave-uniform rescale
    mrun = mnew;
#pragma unroll
    for (int r = 0; r < 8; ++r) {
      float p0 = __expf(s0[r] - mnew);
      float p1 = __expf(s1[r] - mnew);
      int prow = r + hi * 8;
      lp[wave][prow * 32 + cl]      = (_Float16)p0;
      lp[wave][prow * 32 + cl + 16] = (_Float16)p1;
      o0[r] *= alpha;
      o1[r] *= alpha;
      lrun[r] *= alpha;
    }
    // Wave-private LDS round trip: C-layout -> A-frag (DS ops are in-order).
    v16h pf = load_a(&lp[wave][0], 32);
    lrun = wmma_f16(pf, onesb, lrun);        // += rowsum(P), no shuffles
    o0 = wmma_f16(pf, vf0, o0);
    o1 = wmma_f16(pf, vf1, o1);

    kf0 = nkf0; kf1 = nkf1; vf0 = nvf0; vf1 = nvf1;
#pragma unroll
    for (int r = 0; r < 8; ++r) mw[r] = nmw[r];
  }
#pragma unroll
  for (int r = 0; r < 8; ++r) {              // normalize + write [B,G,D]
    int g_ = q0 + r + hi * 8;
    float inv = 1.0f / lrun[r];
    _Float16* orow = o + ((size_t)b_ * G_ + g_) * D_ + h_ * DH_;
    orow[cl]      = (_Float16)(o0[r] * inv);
    orow[cl + 16] = (_Float16)(o1[r] * inv);
  }
}

// ---------------------------------------------------------------- pool + head
__global__ __launch_bounds__(256) void pool_kernel(const float* __restrict__ x,
                                                   float* __restrict__ emb) {
  int t = blockIdx.x * 256 + threadIdx.x;    // < B*D
  int b = t >> 7, d = t & (D_ - 1);
  const float* p = x + (size_t)b * G_ * D_ + d;
  float s = 0.0f;
  for (int g = 0; g < G_; ++g) s += p[(size_t)g * D_];
  emb[t] = s * (1.0f / G_);
}

__global__ __launch_bounds__(256) void head_kernel(
    const float* __restrict__ emb, const float* __restrict__ ng,
    const float* __restrict__ nb, const float* __restrict__ cw,
    const float* __restrict__ cb, float* __restrict__ out) {
  __shared__ float se[B_ * D_];
  int wave = threadIdx.x >> 5, lane = threadIdx.x & 31;
  const float* er = emb + wave * D_;         // wave b handles row b (8 waves)
  v4f v = *(const v4f*)(er + lane * 4);
  float s = v[0] + v[1] + v[2] + v[3];
  for (int off = 16; off; off >>= 1) s += __shfl_xor(s, off, 32);
  float mu = s * (1.0f / D_);
  float d0 = v[0] - mu, d1 = v[1] - mu, d2 = v[2] - mu, d3 = v[3] - mu;
  float q = d0 * d0 + d1 * d1 + d2 * d2 + d3 * d3;
  for (int off = 16; off; off >>= 1) q += __shfl_xor(q, off, 32);
  float rs = rsqrtf(q * (1.0f / D_) + 1e-5f);
  int c = lane * 4;
  float l0 = d0 * rs * ng[c + 0] + nb[c + 0];
  float l1 = d1 * rs * ng[c + 1] + nb[c + 1];
  float l2 = d2 * rs * ng[c + 2] + nb[c + 2];
  float l3 = d3 * rs * ng[c + 3] + nb[c + 3];
  se[wave * D_ + c + 0] = l0;  out[B_ * NCLS_ + wave * D_ + c + 0] = l0;
  se[wave * D_ + c + 1] = l1;  out[B_ * NCLS_ + wave * D_ + c + 1] = l1;
  se[wave * D_ + c + 2] = l2;  out[B_ * NCLS_ + wave * D_ + c + 2] = l2;
  se[wave * D_ + c + 3] = l3;  out[B_ * NCLS_ + wave * D_ + c + 3] = l3;
  __syncthreads();
  if (threadIdx.x < B_ * NCLS_) {            // logits: 160 dot products
    int b = threadIdx.x / NCLS_, cc = threadIdx.x % NCLS_;
    float a = cb[cc];
    for (int d = 0; d < D_; ++d) a += se[b * D_ + d] * cw[d * NCLS_ + cc];
    out[b * NCLS_ + cc] = a;
  }
}

// ---------------------------------------------------------------- workspace
constexpr size_t SZ_XF32 = (size_t)BG_ * D_ * 4;   // 8 MiB
constexpr size_t SZ_ACT  = (size_t)BG_ * D_ * 2;   // 4 MiB f16
constexpr size_t OFF_X  = 0;
constexpr size_t OFF_H  = OFF_X + SZ_XF32;
constexpr size_t OFF_Q  = OFF_H + SZ_ACT;
constexpr size_t OFF_KT = OFF_Q + SZ_ACT;
constexpr size_t OFF_V  = OFF_KT + SZ_ACT;
constexpr size_t OFF_O  = OFF_V + SZ_ACT;
constexpr size_t OFF_MH = OFF_Q;                   // MLP hidden aliases Q..O
constexpr size_t OFF_WQ = OFF_O + SZ_ACT;          // 28 MiB, f16 weight copies
constexpr size_t OFF_WK = OFF_WQ + (size_t)D_ * D_ * 2;
constexpr size_t OFF_WV = OFF_WK + (size_t)D_ * D_ * 2;
constexpr size_t OFF_WO = OFF_WV + (size_t)D_ * D_ * 2;
constexpr size_t OFF_W1 = OFF_WO + (size_t)D_ * D_ * 2;
constexpr size_t OFF_W2 = OFF_W1 + (size_t)D_ * DM_ * 2;
constexpr size_t OFF_EM = OFF_W2 + (size_t)DM_ * D_ * 2;
constexpr size_t OFF_PM = OFF_EM + (size_t)B_ * D_ * 4;   // packed mask 512KB

extern "C" void kernel_launch(void* const* d_in, const int* in_sizes, int n_in,
                              void* d_out, int out_size, void* d_ws,
                              size_t ws_size, hipStream_t stream) {
  (void)in_sizes; (void)n_in; (void)out_size; (void)ws_size;
  auto in = [&](int i) { return (const float*)d_in[i]; };
  const float* values = in(0);
  const unsigned char* mask = (const unsigned char*)d_in[1];
  const float* gene_embed = in(2);
  const float* val_w = in(3);
  const float* val_b = in(4);
  const float* norm_g = in(37);
  const float* norm_b = in(38);
  const float* clf_w = in(39);
  const float* clf_b = in(40);

  char* ws = (char*)d_ws;
  float*    x   = (float*)(ws + OFF_X);
  _Float16* h   = (_Float16*)(ws + OFF_H);
  _Float16* qb  = (_Float16*)(ws + OFF_Q);
  _Float16* ktb = (_Float16*)(ws + OFF_KT);
  _Float16* vb  = (_Float16*)(ws + OFF_V);
  _Float16* ob  = (_Float16*)(ws + OFF_O);
  _Float16* mh  = (_Float16*)(ws + OFF_MH);
  _Float16* wqh = (_Float16*)(ws + OFF_WQ);
  _Float16* wkh = (_Float16*)(ws + OFF_WK);
  _Float16* wvh = (_Float16*)(ws + OFF_WV);
  _Float16* woh = (_Float16*)(ws + OFF_WO);
  _Float16* w1h = (_Float16*)(ws + OFF_W1);
  _Float16* w2h = (_Float16*)(ws + OFF_W2);
  float*        emb = (float*)(ws + OFF_EM);
  unsigned int* pm  = (unsigned int*)(ws + OFF_PM);

  // One-time-per-launch prep: bit-pack the mask; embed the inputs.
  packmask_kernel<<<G_ * G_ / 256, 256, 0, stream>>>(mask, pm);
  embed_kernel<<<BG_ * D_ / 256, 256, 0, stream>>>(values, gene_embed, val_w,
                                                   val_b, x);

  for (int l = 0; l < 2; ++l) {
    int base = 5 + l * 16;   // ln1_g, ln1_b, wq,bq, wk,bk, wv,bv, wo,bo,
                             // ln2_g, ln2_b, w1,b1, w2,b2
    const float *ln1_g = in(base + 0), *ln1_b = in(base + 1);
    const float *wq = in(base + 2), *bq = in(base + 3);
    const float *wk = in(base + 4), *bk = in(base + 5);
    const float *wv = in(base + 6), *bv = in(base + 7);
    const float *wo = in(base + 8), *bo = in(base + 9);
    const float *ln2_g = in(base + 10), *ln2_b = in(base + 11);
    const float *w1 = in(base + 12), *b1 = in(base + 13);
    const float *w2 = in(base + 14), *b2 = in(base + 15);

    cast_kernel<<<(D_ * D_ + 255) / 256, 256, 0, stream>>>(wq, wqh, D_ * D_);
    cast_kernel<<<(D_ * D_ + 255) / 256, 256, 0, stream>>>(wk, wkh, D_ * D_);
    cast_kernel<<<(D_ * D_ + 255) / 256, 256, 0, stream>>>(wv, wvh, D_ * D_);
    cast_kernel<<<(D_ * D_ + 255) / 256, 256, 0, stream>>>(wo, woh, D_ * D_);
    cast_kernel<<<(D_ * DM_ + 255) / 256, 256, 0, stream>>>(w1, w1h, D_ * DM_);
    cast_kernel<<<(DM_ * D_ + 255) / 256, 256, 0, stream>>>(w2, w2h, DM_ * D_);

    ln_kernel<<<BG_ / 8, 256, 0, stream>>>(x, ln1_g, ln1_b, h);

    // Q,K,V projections (16384x128 @ 128x128), K stored transposed.
    gemm_kernel<EPI_QV, D_, D_><<<256, 256, 0, stream>>>(h, wqh, bq, nullptr, qb);
    gemm_kernel<EPI_KT, D_, D_><<<256, 256, 0, stream>>>(h, wkh, bk, nullptr, ktb);
    gemm_kernel<EPI_QV, D_, D_><<<256, 256, 0, stream>>>(h, wvh, bv, nullptr, vb);

    attn_kernel<<<(B_ * H_ * (G_ / 16)) / 8, 256, 0, stream>>>(qb, ktb, vb, pm,
                                                               ob);

    // x += o @ wo + bo
    gemm_kernel<EPI_ADD_RES, D_, D_><<<256, 256, 0, stream>>>(ob, woh, bo, x,
                                                              nullptr);

    ln_kernel<<<BG_ / 8, 256, 0, stream>>>(x, ln2_g, ln2_b, h);

    // MLP: gelu(h @ w1 + b1) @ w2 + b2, residual into x
    gemm_kernel<EPI_GELU_F16, DM_, D_><<<1024, 256, 0, stream>>>(h, w1h, b1,
                                                                 nullptr, mh);
    gemm_kernel<EPI_ADD_RES, D_, DM_><<<256, 256, 0, stream>>>(mh, w2h, b2, x,
                                                               nullptr);
  }

  pool_kernel<<<(B_ * D_) / 256, 256, 0, stream>>>(x, emb);
  head_kernel<<<1, 256, 0, stream>>>(emb, norm_g, norm_b, clf_w, clf_b,
                                     (float*)d_out);
}
